// CNN4d_4088808866210
// MI455X (gfx1250) — matrix operand order repaired and checked by reading, not compiled, not run on hardware
//
#include <hip/hip_runtime.h>
#include <hip/hip_bf16.h>

// ---------------- problem constants ----------------
#define T_ 6
#define X_ 24
#define Y_ 24
#define Z_ 12
#define NPTS (T_ * X_ * Y_ * Z_)   // 41472
#define NTAPS 625                  // 5^4
#define NBLK_STATS 64

// padded volume (halo of 2 on every side of all 4 dims)
#define PD_T 10
#define PD_X 28
#define PD_Y 28
#define PD_Z 16
#define PSZ_Y PD_Z                  // y stride in rows = 16
#define PSZ_X (PD_Y * PD_Z)         // x stride = 448
#define PSZ_T (PD_X * PD_Y * PD_Z)  // t stride = 12544
#define NPAD (PD_T * PSZ_T)         // 125440 padded rows

typedef __attribute__((ext_vector_type(16))) __bf16 v16bf;
typedef __attribute__((ext_vector_type(8)))  float  v8f;

union ABu { uint4 u[2]; v16bf v; };
union Du  { v8f v; float4 q[2]; };

__device__ __forceinline__ int padded_row(int t, int x, int y, int z) {
    return (t + 2) * PSZ_T + (x + 2) * PSZ_X + (y + 2) * PSZ_Y + (z + 2);
}

// ---------------- workspace layout (bytes) ----------------
static constexpr size_t SZ_ACTP = (size_t)NPAD * 32 * 2;      // 8,028,160 (max C=32, bf16)
static constexpr size_t SZ_CONV = (size_t)NPTS * 32 * 4;      // 5,308,416
static constexpr size_t SZ_PART = 16384;
static constexpr size_t SZ_ST   = 1024;
static constexpr size_t SZ_TAB  = 2560;                        // 626 ints, padded
static constexpr size_t SZ_A1   = (size_t)313 * 1 * 32 * 32;
static constexpr size_t SZ_A2   = (size_t)313 * 2 * 32 * 32;
static constexpr size_t OFF_BUFA = 0;
static constexpr size_t OFF_BUFB = OFF_BUFA + SZ_ACTP;
static constexpr size_t OFF_CONV = OFF_BUFB + SZ_ACTP;
static constexpr size_t OFF_PART = OFF_CONV + SZ_CONV;
static constexpr size_t OFF_ST   = OFF_PART + SZ_PART;
static constexpr size_t OFF_TAB  = OFF_ST + SZ_ST;
static constexpr size_t OFF_A1   = OFF_TAB + SZ_TAB;
static constexpr size_t OFF_A2   = OFF_A1 + SZ_A1;
static constexpr size_t OFF_A3   = OFF_A2 + SZ_A2;

// ---------------- zero padded activation buffers ----------------
__global__ void zero_kernel(uint4* __restrict__ p, int n16) {
    uint4 z; z.x = 0; z.y = 0; z.z = 0; z.w = 0;
    for (int i = blockIdx.x * blockDim.x + threadIdx.x; i < n16; i += gridDim.x * blockDim.x)
        p[i] = z;
}

// ---------------- tap table: flat padded-row shift per tap ----------------
__global__ void taptab_kernel(int* __restrict__ tab) {
    int i = blockIdx.x * blockDim.x + threadIdx.x;
    if (i >= 626) return;
    unsigned ut = (unsigned)(i < NTAPS ? i : NTAPS - 1);  // pad tap aliases 624 (A is zero there)
    int dz = (int)(ut % 5u) - 2; ut /= 5u;
    int dy = (int)(ut % 5u) - 2; ut /= 5u;
    int dx = (int)(ut % 5u) - 2;
    int dt = (int)(ut / 5u) - 2;
    tab[i] = dt * PSZ_T + dx * PSZ_X + dy * PSZ_Y + dz;
}

// ---------------- weight repack into WMMA A-fragment layout ----------------
// apk layout: [kc][ct][lane][16 bf16]; A 16x32 bf16, this code's contiguous k order:
// lane<16 holds K=0..15 ; lane>=16 holds K=16..31
__global__ void repackA(const float* __restrict__ w, __hip_bfloat16* __restrict__ apk,
                        int CIN, int COT) {
    int KCH = (CIN == 16) ? 313 : 625;
    int id = blockIdx.x * blockDim.x + threadIdx.x;
    if (id >= KCH * COT * 32) return;
    int lane = id & 31;
    int ct   = (id >> 5) % COT;
    int kc   = id / (32 * COT);
    int half = lane >> 4;
    int row  = (lane & 15) + ct * 16;
    __hip_bfloat16* dst = apk + (size_t)id * 16;
#pragma unroll
    for (int j = 0; j < 16; ++j) {
        int k = j + half * 16;   // K index 0..31
        int cin, tap;
        if (CIN == 16) { tap = kc * 2 + (k >> 4); cin = k & 15; }
        else           { tap = kc;                cin = k; }
        float v = (tap < NTAPS) ? w[((size_t)row * CIN + cin) * NTAPS + tap] : 0.f;
        dst[j] = __float2bfloat16(v);
    }
}

// ---------------- embedding -> padded bf16 act (prow,16) ----------------
__global__ __launch_bounds__(256) void embed_kernel(const float* __restrict__ x,
                                                    const float* __restrict__ wemb,
                                                    const float* __restrict__ bemb,
                                                    __hip_bfloat16* __restrict__ out) {
    int p = blockIdx.x * blockDim.x + threadIdx.x;
    if (p >= NPTS) return;
    int z = p % Z_; int r = p / Z_;
    int y = r % Y_; r /= Y_;
    int xx = r % X_; int t = r / X_;
    size_t prow = (size_t)padded_row(t, xx, y, z);
    const float4* xp = reinterpret_cast<const float4*>(x + (size_t)p * 8);
    float4 x0 = xp[0], x1 = xp[1];
    float xv[8] = {x0.x, x0.y, x0.z, x0.w, x1.x, x1.y, x1.z, x1.w};
#pragma unroll
    for (int c = 0; c < 16; ++c) {
        float h = bemb[c];
#pragma unroll
        for (int f = 0; f < 8; ++f) h += xv[f] * wemb[f * 16 + c];
        out[prow * 16 + c] = __float2bfloat16(h);
    }
}

// ---------------- 4D conv via implicit-GEMM WMMA, K-split x2 + LDS combine ----------------
// act: padded bf16 [NPAD x CIN] (halo rows are zero) ; out: f32 compact [NPTS x C]
// block = 8 waves = 4 tile-groups x 2 K-halves; each wave: PT=2 point tiles.
template <int CIN, int COT>
__global__ __launch_bounds__(256) void conv4d_wmma(const __hip_bfloat16* __restrict__ act,
                                                   const __hip_bfloat16* __restrict__ apk,
                                                   const int* __restrict__ taptab,
                                                   float* __restrict__ out) {
    constexpr int KCH  = (CIN == 16) ? 313 : 625;  // K=32 chunks
    constexpr int PT   = 2;                         // point tiles per wave
    constexpr int GPB  = 4;                         // tile-groups per block
    constexpr int C    = COT * 16;
    constexpr int ROWB = CIN * 2;                   // bytes per padded row

    __shared__ int   s_tab[626];
    __shared__ float s_red[GPB * PT * COT * 256];   // partial D matrices (K-half 1)

    for (int i = threadIdx.x; i < 626; i += 256) s_tab[i] = taptab[i] * ROWB; // pre-scaled
    __syncthreads();

    const int lane  = threadIdx.x & 31;
    const int wave  = threadIdx.x >> 5;             // 0..7
    const int g     = wave >> 1;                    // tile-group in block
    const int khalf = wave & 1;                     // K-split half
    const int half  = lane >> 4;
    const int n     = lane & 15;
    const int tile0 = (blockIdx.x * GPB + g) * PT;

    int bp[PT];     // compact point index (store)
    int pbase[PT];  // byte offset of this lane's B row in padded act
#pragma unroll
    for (int i = 0; i < PT; ++i) {
        int p = (tile0 + i) * 16 + n;
        bp[i] = p;
        int z = p % Z_; int r = p / Z_;
        int y = r % Y_; r /= Y_;
        int xx = r % X_; int t = r / X_;
        pbase[i] = padded_row(t, xx, y, z) * ROWB + (CIN == 32 ? half * 32 : 0);
    }

    v8f zero8 = {0.f, 0.f, 0.f, 0.f, 0.f, 0.f, 0.f, 0.f};
    v8f acc[PT][COT];
#pragma unroll
    for (int i = 0; i < PT; ++i)
#pragma unroll
        for (int ct = 0; ct < COT; ++ct) acc[i][ct] = zero8;

    const uint4* apk4 = reinterpret_cast<const uint4*>(apk);
    const char*  actc = reinterpret_cast<const char*>(act);

    constexpr int KH = (KCH + 1) / 2;
    const int kc0 = khalf ? KH : 0;
    const int kc1 = khalf ? KCH : KH;

    for (int kc = kc0; kc < kc1; ++kc) {
        int tap = (CIN == 16) ? (kc * 2 + half) : kc;  // two taps per chunk when CIN==16
        int sb  = s_tab[tap];                          // byte shift (pre-scaled, may be negative)

        ABu a[COT];
#pragma unroll
        for (int ct = 0; ct < COT; ++ct) {
            size_t o = ((size_t)(kc * COT + ct) * 32 + lane) * 2;
            a[ct].u[0] = apk4[o];
            a[ct].u[1] = apk4[o + 1];
        }
#pragma unroll
        for (int i = 0; i < PT; ++i) {
            const uint4* b4 = reinterpret_cast<const uint4*>(actc + (pbase[i] + sb));
            ABu b;
            b.u[0] = b4[0];
            b.u[1] = b4[1];
#pragma unroll
            for (int ct = 0; ct < COT; ++ct)
                acc[i][ct] = __builtin_amdgcn_wmma_f32_16x16x32_bf16(
                    false, a[ct].v, false, b.v, (short)0, acc[i][ct], false, false);
        }
    }

    // deterministic K-half combine through LDS (fixed pairing, fixed order)
    if (khalf == 1) {
#pragma unroll
        for (int i = 0; i < PT; ++i)
#pragma unroll
            for (int ct = 0; ct < COT; ++ct) {
                float4* d = reinterpret_cast<float4*>(
                    &s_red[(((g * PT + i) * COT + ct) * 32 + lane) * 8]);
                Du dd; dd.v = acc[i][ct];
                d[0] = dd.q[0];
                d[1] = dd.q[1];
            }
    }
    __syncthreads();
    if (khalf == 0) {
#pragma unroll
        for (int i = 0; i < PT; ++i)
#pragma unroll
            for (int ct = 0; ct < COT; ++ct) {
                const float4* d = reinterpret_cast<const float4*>(
                    &s_red[(((g * PT + i) * COT + ct) * 32 + lane) * 8]);
                Du o; o.q[0] = d[0]; o.q[1] = d[1];
                v8f r = acc[i][ct] + o.v;
                // D layout: lane<16 -> N=lane, M=v ; lane>=16 -> N=lane-16, M=8+v
                Du dd; dd.v = r;
                float4* o4 = reinterpret_cast<float4*>(
                    out + (size_t)bp[i] * C + ct * 16 + half * 8);
                o4[0] = dd.q[0];
                o4[1] = dd.q[1];
            }
    }
}

// ---------------- deterministic BN statistics (two stage) ----------------
template <int C>
__global__ __launch_bounds__(256) void stats_kernel(const float* __restrict__ conv,
                                                    float* __restrict__ partial) {
    __shared__ float sh[512];
    int tid = threadIdx.x;
    int c = tid & (C - 1);
    int row = (blockIdx.x * 256 + tid) / C;
    int rstride = (gridDim.x * 256) / C;
    float s = 0.f, q = 0.f;
    for (int p = row; p < NPTS; p += rstride) {
        float v = conv[(size_t)p * C + c];
        s += v; q += v * v;
    }
    sh[tid] = s; sh[256 + tid] = q;
    __syncthreads();
    if (tid < C) {
        float ts = 0.f, tq = 0.f;
        for (int t = tid; t < 256; t += C) { ts += sh[t]; tq += sh[256 + t]; }
        partial[blockIdx.x * 2 * C + tid]     = ts;
        partial[blockIdx.x * 2 * C + C + tid] = tq;
    }
}

__global__ void stats_reduce(const float* __restrict__ partial, float* __restrict__ st,
                             int C, int nblk) {
    int j = threadIdx.x;
    if (j < 2 * C) {
        float s = 0.f;
        for (int b = 0; b < nblk; ++b) s += partial[b * 2 * C + j];
        st[j] = s;
    }
}

// ---------------- BN (training stats) + LeakyReLU -> padded bf16 ----------------
template <int C>
__global__ __launch_bounds__(256) void bn_lrelu_kernel(const float* __restrict__ conv,
                                                       const float* __restrict__ st,
                                                       const float* __restrict__ gamma,
                                                       const float* __restrict__ beta,
                                                       __hip_bfloat16* __restrict__ out) {
    int idx = blockIdx.x * blockDim.x + threadIdx.x;
    if (idx >= NPTS * C) return;
    int c = idx & (C - 1);
    int p = idx / C;
    int z = p % Z_; int r = p / Z_;
    int y = r % Y_; r /= Y_;
    int xx = r % X_; int t = r / X_;
    size_t prow = (size_t)padded_row(t, xx, y, z);
    const float inv = 1.f / (float)NPTS;
    float mean = st[c] * inv;
    float var  = st[C + c] * inv - mean * mean;
    float sc = gamma[c] * rsqrtf(var + 1e-5f);
    float sh = beta[c] - mean * sc;
    float v = conv[idx] * sc + sh;
    v = (v >= 0.f) ? v : 0.1f * v;
    out[prow * C + c] = __float2bfloat16(v);
}

// ---------------- final projection ----------------
__global__ __launch_bounds__(256) void proj_kernel(const __hip_bfloat16* __restrict__ act,
                                                   const float* __restrict__ wp,
                                                   const float* __restrict__ bpj,
                                                   float* __restrict__ out) {
    int p = blockIdx.x * blockDim.x + threadIdx.x;
    if (p >= NPTS) return;
    int z = p % Z_; int r = p / Z_;
    int y = r % Y_; r /= Y_;
    int xx = r % X_; int t = r / X_;
    size_t prow = (size_t)padded_row(t, xx, y, z);
    float s = bpj[0];
#pragma unroll
    for (int c = 0; c < 16; ++c)
        s += __bfloat162float(act[prow * 16 + c]) * wp[c];
    out[p] = s;
}

// ---------------- host-side orchestration ----------------
extern "C" void kernel_launch(void* const* d_in, const int* in_sizes, int n_in,
                              void* d_out, int out_size, void* d_ws, size_t ws_size,
                              hipStream_t stream) {
    (void)in_sizes; (void)n_in; (void)out_size; (void)ws_size;
    const float* x      = (const float*)d_in[0];
    const float* w_emb  = (const float*)d_in[1];
    const float* b_emb  = (const float*)d_in[2];
    const float* w1     = (const float*)d_in[3];
    // d_in[4]/[8]/[12] (conv biases) cancel under training-mode batchnorm
    const float* g1     = (const float*)d_in[5];
    const float* be1    = (const float*)d_in[6];
    const float* w2     = (const float*)d_in[7];
    const float* g2     = (const float*)d_in[9];
    const float* be2    = (const float*)d_in[10];
    const float* w3     = (const float*)d_in[11];
    const float* g3     = (const float*)d_in[13];
    const float* be3    = (const float*)d_in[14];
    const float* w_proj = (const float*)d_in[15];
    const float* b_proj = (const float*)d_in[16];
    float* out = (float*)d_out;

    char* ws = (char*)d_ws;
    __hip_bfloat16* bufA = (__hip_bfloat16*)(ws + OFF_BUFA);
    __hip_bfloat16* bufB = (__hip_bfloat16*)(ws + OFF_BUFB);
    float* convout = (float*)(ws + OFF_CONV);
    float* partial = (float*)(ws + OFF_PART);
    float* st1 = (float*)(ws + OFF_ST);
    float* st2 = st1 + 64;
    float* st3 = st1 + 128;
    int*   tab = (int*)(ws + OFF_TAB);
    __hip_bfloat16* A1 = (__hip_bfloat16*)(ws + OFF_A1);
    __hip_bfloat16* A2 = (__hip_bfloat16*)(ws + OFF_A2);
    __hip_bfloat16* A3 = (__hip_bfloat16*)(ws + OFF_A3);

    // zero both padded activation buffers (halo must be zero)
    int n16 = (int)((2 * SZ_ACTP) / 16);
    zero_kernel<<<512, 256, 0, stream>>>((uint4*)ws, n16);

    taptab_kernel<<<3, 256, 0, stream>>>(tab);
    repackA<<<(313 * 1 * 32 + 255) / 256, 256, 0, stream>>>(w1, A1, 16, 1);
    repackA<<<(313 * 2 * 32 + 255) / 256, 256, 0, stream>>>(w2, A2, 16, 2);
    repackA<<<(625 * 1 * 32 + 255) / 256, 256, 0, stream>>>(w3, A3, 32, 1);

    embed_kernel<<<NPTS / 256, 256, 0, stream>>>(x, w_emb, b_emb, bufA);

    // 2592 point tiles, 8 tiles per block (4 groups x PT=2) -> 324 blocks, K-split x2
    // layer 1: 16 -> 16
    conv4d_wmma<16, 1><<<324, 256, 0, stream>>>(bufA, A1, tab, convout);
    stats_kernel<16><<<NBLK_STATS, 256, 0, stream>>>(convout, partial);
    stats_reduce<<<1, 64, 0, stream>>>(partial, st1, 16, NBLK_STATS);
    bn_lrelu_kernel<16><<<NPTS * 16 / 256, 256, 0, stream>>>(convout, st1, g1, be1, bufB);

    // layer 2: 16 -> 32
    conv4d_wmma<16, 2><<<324, 256, 0, stream>>>(bufB, A2, tab, convout);
    stats_kernel<32><<<NBLK_STATS, 256, 0, stream>>>(convout, partial);
    stats_reduce<<<1, 64, 0, stream>>>(partial, st2, 32, NBLK_STATS);
    bn_lrelu_kernel<32><<<NPTS * 32 / 256, 256, 0, stream>>>(convout, st2, g2, be2, bufA);

    // layer 3: 32 -> 16
    conv4d_wmma<32, 1><<<324, 256, 0, stream>>>(bufA, A3, tab, convout);
    stats_kernel<16><<<NBLK_STATS, 256, 0, stream>>>(convout, partial);
    stats_reduce<<<1, 64, 0, stream>>>(partial, st3, 16, NBLK_STATS);
    bn_lrelu_kernel<16><<<NPTS * 16 / 256, 256, 0, stream>>>(convout, st3, g3, be3, bufB);

    proj_kernel<<<NPTS / 256, 256, 0, stream>>>(bufB, w_proj, b_proj, out);
}
